// InversionModel_26474178413226
// MI455X (gfx1250) — matrix operand order repaired and checked by reading, not compiled, not run on hardware
//
#include <hip/hip_runtime.h>

// ---------------------------------------------------------------------------
// Types for CDNA5 WMMA (wave32)
// ---------------------------------------------------------------------------
typedef __attribute__((ext_vector_type(16))) __bf16 bf16x16;
typedef __attribute__((ext_vector_type(8)))  __bf16 bf16x8;
typedef __attribute__((ext_vector_type(8)))  float  f32x8;

#define M_DIM 2048
#define K_DIM 6432
#define N_DIM 6432

#define KSTEP   96                  // 6432 = 67 * 96 (no tail)
#define NT      (K_DIM / KSTEP)     // 67 iterations
#define B_LDSTR 104                 // padded row stride (elems): 104*2B/4 = 52
                                    // banks/row -> 16 rows hit distinct
                                    // multiples-of-4 banks: conflict-free b128

// ---------------------------------------------------------------------------
// gfx1250 async global->LDS copy (ASYNCcnt engine).  Inline asm per the
// bridge doc: portable across ROCm-7.2 / amdgpu-toolchain builtins.
// ---------------------------------------------------------------------------
__device__ __forceinline__ void async_load_b128(const __bf16* g, __bf16* l) {
    unsigned lofs = (unsigned)(unsigned long long)
        (__attribute__((address_space(3))) const __bf16*)l;
    asm volatile("global_load_async_to_lds_b128 %0, %1, off"
                 :: "v"(lofs), "v"(g) : "memory");
}
__device__ __forceinline__ void wait_async0() {
    asm volatile("s_wait_asynccnt 0x0" ::: "memory");
}

// ---------------------------------------------------------------------------
// 1) f32 -> bf16 weight conversion (wl becomes L2-resident bf16, 83 MB)
// ---------------------------------------------------------------------------
__global__ __launch_bounds__(256) void f32_to_bf16_kernel(
    const float* __restrict__ in, __bf16* __restrict__ out, long n)
{
    long i = (long)blockIdx.x * 256 + threadIdx.x;
    long stride = (long)gridDim.x * 256;
    for (; i < n; i += stride)
        out[i] = (__bf16)in[i];
}

// ---------------------------------------------------------------------------
// 2) Multi-scale conv front-end.  One block per batch element, all stages in
//    LDS.  Emits feature row f[b, 0:6432] in bf16, flat layout c2*201 + pos.
// ---------------------------------------------------------------------------
__global__ __launch_bounds__(256) void conv_features_kernel(
    const float* __restrict__ x,   // [2048, 4, 480]
    const float* __restrict__ w1,  // [16, 4, 5]
    const float* __restrict__ b1,  // [16]
    const float* __restrict__ w2,  // [32, 16, 5]
    const float* __restrict__ b2,  // [32]
    __bf16* __restrict__ feat)     // [2048, 6432] bf16
{
    __shared__ float sx [4 * 480];
    __shared__ float sxs[4 * 480];
    __shared__ float sh1[16 * 238];
    __shared__ float sw1[16 * 4 * 5];
    __shared__ float sb1[16];
    __shared__ float sw2[32 * 16 * 5];
    __shared__ float sb2[32];

    const int b = blockIdx.x;
    const int t = threadIdx.x;

    for (int i = t; i < 4 * 480; i += 256) sx[i] = x[(size_t)b * (4 * 480) + i];
    for (int i = t; i < 16 * 4 * 5; i += 256) sw1[i] = w1[i];
    for (int i = t; i < 32 * 16 * 5; i += 256) sw2[i] = w2[i];
    if (t < 16) sb1[t] = b1[t];
    if (t < 32) sb2[t] = b2[t];
    __syncthreads();

    const int scales[3] = {1, 2, 4};
    const int offs[3]   = {0, 117, 174};

    for (int si = 0; si < 3; ++si) {
        const int s  = scales[si];
        const int Ls = 480 / s;            // 480, 240, 120
        const int P1 = (Ls - 4) / 2;       // 238, 118, 58
        const int P2 = (P1 - 4) / 2;       // 117, 57, 27

        for (int i = t; i < 4 * Ls; i += 256) {
            const int c = i / Ls, l = i % Ls;
            float acc = 0.f;
            for (int u = 0; u < s; ++u) acc += sx[c * 480 + l * s + u];
            sxs[c * 480 + l] = acc * (1.0f / (float)s);
        }
        __syncthreads();

        for (int i = t; i < 16 * P1; i += 256) {
            const int c1 = i / P1, p = i % P1;
            float v0 = sb1[c1], v1 = sb1[c1];
            for (int c0 = 0; c0 < 4; ++c0) {
                #pragma unroll
                for (int k = 0; k < 5; ++k) {
                    const float w = sw1[(c1 * 4 + c0) * 5 + k];
                    v0 += sxs[c0 * 480 + 2 * p + k] * w;
                    v1 += sxs[c0 * 480 + 2 * p + 1 + k] * w;
                }
            }
            float v = v0 > v1 ? v0 : v1;
            sh1[c1 * 238 + p] = v > 0.f ? v : 0.f;
        }
        __syncthreads();

        for (int i = t; i < 32 * P2; i += 256) {
            const int c2 = i / P2, p = i % P2;
            float v0 = sb2[c2], v1 = sb2[c2];
            for (int c1 = 0; c1 < 16; ++c1) {
                #pragma unroll
                for (int k = 0; k < 5; ++k) {
                    const float w = sw2[(c2 * 16 + c1) * 5 + k];
                    v0 += sh1[c1 * 238 + 2 * p + k] * w;
                    v1 += sh1[c1 * 238 + 2 * p + 1 + k] * w;
                }
            }
            float v = v0 > v1 ? v0 : v1;
            v = v > 0.f ? v : 0.f;
            feat[(size_t)b * N_DIM + c2 * 201 + offs[si] + p] = (__bf16)v;
        }
        __syncthreads();
    }
}

// ---------------------------------------------------------------------------
// 3) WMMA GEMM:  Out = relu(A[2048,6432] * W[6432,6432]^T + bias).
//    Block tile 128(M) x 96(N), 8 waves, wave tile 32x48.
//    B tile (shared by all 4 M-waves) is staged via async global->LDS,
//    double-buffered over a 96-wide K step; A fragments load directly from
//    global (K-contiguous b128 per lane, WGP$/L2 resident).
// ---------------------------------------------------------------------------
__global__ __launch_bounds__(256) void gemm_relu_bf16_kernel(
    const __bf16* __restrict__ A,     // [2048, 6432] row-major (K contiguous)
    const __bf16* __restrict__ W,     // [6432, 6432] row-major (K contiguous)
    const float*  __restrict__ bias,  // [6432]
    __bf16* __restrict__ Out)         // [2048, 6432]
{
    __shared__ __bf16 sB[2][96 * B_LDSTR];   // ~39 KB double-buffered B tile

    const int lane = threadIdx.x & 31;
    const int wave = threadIdx.x >> 5;
    const int h    = lane >> 4;       // K half selector (ISA bf16 layout)
    const int mr   = lane & 15;
    const int wm   = wave & 3;        // 4 waves along M
    const int wn   = wave >> 2;       // 2 waves along N

    const int mBase  = blockIdx.y * 128 + wm * 32;
    const int nBlock = blockIdx.x * 96;
    const int nBase  = nBlock + wn * 48;

    // --- A fragment pointers (direct global, K-contiguous) ---
    const __bf16* aP[2];
    aP[0] = A + (size_t)(mBase +  0 + mr) * K_DIM + h * 8;
    aP[1] = A + (size_t)(mBase + 16 + mr) * K_DIM + h * 8;

    // --- async B-tile copy: 96 rows x 96 K-elems = 1152 x 16B chunks,
    //     144 chunks per wave (12 chunks per row) ---
    auto issueB = [&](int kt, int bu) {
        #pragma unroll
        for (int i = 0; i < 5; ++i) {
            const int c = wave * 144 + i * 32 + lane;
            if (i < 4 || lane < 16) {            // last instr: half-EXEC
                const int row = c / 12;
                const int cc  = c - row * 12;
                const __bf16* g = W + (size_t)(nBlock + row) * K_DIM + kt + cc * 8;
                async_load_b128(g, &sB[bu][row * B_LDSTR + cc * 8]);
            }
        }
    };

    f32x8 acc[2][3] = {};

    issueB(0, 0);

    for (int it = 0; it < NT; ++it) {
        const int cur = it & 1;
        const int kt  = it * KSTEP;

        wait_async0();        // batch `it` landed in sB[cur]
        __syncthreads();      // visible to all waves; prior readers drained

        if (it + 1 < NT) issueB(kt + KSTEP, cur ^ 1);   // overlap next batch

        // prefetch next A K-step into WGP$ (locality 3 -> WGP scope)
        __builtin_prefetch((const void*)(aP[0] + kt + KSTEP), 0, 3);
        __builtin_prefetch((const void*)(aP[1] + kt + KSTEP), 0, 3);

        #pragma unroll
        for (int s3 = 0; s3 < 3; ++s3) {
            const int kk = s3 * 32;

            bf16x16 afrag[2];
            #pragma unroll
            for (int i = 0; i < 2; ++i) {
                bf16x8 lo = *(const bf16x8*)(aP[i] + kt + kk);      // K=h*8+0..7
                bf16x8 hi = *(const bf16x8*)(aP[i] + kt + kk + 16); // K=16+h*8..
                afrag[i] = __builtin_shufflevector(lo, hi,
                    0, 1, 2, 3, 4, 5, 6, 7, 8, 9, 10, 11, 12, 13, 14, 15);
            }
            bf16x16 bfrag[3];
            #pragma unroll
            for (int j = 0; j < 3; ++j) {
                const __bf16* p =
                    &sB[cur][(wn * 48 + j * 16 + mr) * B_LDSTR + kk + h * 8];
                bf16x8 lo = *(const bf16x8*)(p);        // K = h*8 + 0..7
                bf16x8 hi = *(const bf16x8*)(p + 16);   // K = 16 + h*8 + 0..7
                bfrag[j] = __builtin_shufflevector(lo, hi,
                    0, 1, 2, 3, 4, 5, 6, 7, 8, 9, 10, 11, 12, 13, 14, 15);
            }

            #pragma unroll
            for (int i = 0; i < 2; ++i)
                #pragma unroll
                for (int j = 0; j < 3; ++j)
                    acc[i][j] = __builtin_amdgcn_wmma_f32_16x16x32_bf16(
                        false, afrag[i], false, bfrag[j],
                        (short)0, acc[i][j], false, false);
        }
    }

    // Epilogue: bias + relu + bf16 store (C/D layout: row = 8*h + r, col = mr)
    #pragma unroll
    for (int i = 0; i < 2; ++i) {
        #pragma unroll
        for (int j = 0; j < 3; ++j) {
            const int col = nBase + j * 16 + mr;
            const float bb = bias[col];
            #pragma unroll
            for (int r = 0; r < 8; ++r) {
                const int row = mBase + i * 16 + h * 8 + r;
                float v = acc[i][j][r] + bb;
                v = v > 0.f ? v : 0.f;
                Out[(size_t)row * N_DIM + col] = (__bf16)v;
            }
        }
    }
}

// ---------------------------------------------------------------------------
// 4) Output head: out[b, 0:5] = f[b] @ wo^T + bo  (f32)
// ---------------------------------------------------------------------------
__global__ __launch_bounds__(256) void head_kernel(
    const __bf16* __restrict__ act,  // [2048, 6432]
    const float*  __restrict__ wo,   // [5, 6432]
    const float*  __restrict__ bo,   // [5]
    float* __restrict__ out)         // [2048, 5]
{
    __shared__ float red[256];
    const int b = blockIdx.x;
    const int t = threadIdx.x;
    const __bf16* arow = act + (size_t)b * K_DIM;

    float acc[5] = {0.f, 0.f, 0.f, 0.f, 0.f};
    for (int k = t; k < K_DIM; k += 256) {
        const float a = (float)arow[k];
        #pragma unroll
        for (int j = 0; j < 5; ++j) acc[j] += a * wo[j * K_DIM + k];
    }
    for (int j = 0; j < 5; ++j) {
        red[t] = acc[j];
        __syncthreads();
        for (int s = 128; s > 0; s >>= 1) {
            if (t < s) red[t] += red[t + s];
            __syncthreads();
        }
        if (t == 0) out[b * 5 + j] = red[0] + bo[j];
        __syncthreads();
    }
}

// ---------------------------------------------------------------------------
// Host-side orchestration
// ---------------------------------------------------------------------------
static inline size_t align256(size_t x) { return (x + 255) & ~(size_t)255; }

extern "C" void kernel_launch(void* const* d_in, const int* in_sizes, int n_in,
                              void* d_out, int out_size, void* d_ws, size_t ws_size,
                              hipStream_t stream) {
    const float* x  = (const float*)d_in[0];
    const float* w1 = (const float*)d_in[1];
    const float* b1 = (const float*)d_in[2];
    const float* w2 = (const float*)d_in[3];
    const float* b2 = (const float*)d_in[4];
    const float* wl = (const float*)d_in[5];
    const float* bl = (const float*)d_in[6];
    const float* wo = (const float*)d_in[7];
    const float* bo = (const float*)d_in[8];

    char* ws = (char*)d_ws;
    size_t off = 0;
    __bf16* wl_bf16 = (__bf16*)(ws + off);
    off += align256((size_t)K_DIM * N_DIM * sizeof(__bf16));     // 82.7 MB
    __bf16* act0 = (__bf16*)(ws + off);
    off += align256((size_t)M_DIM * N_DIM * sizeof(__bf16));     // 26.3 MB
    __bf16* act1 = (__bf16*)(ws + off);                          // 26.3 MB
    (void)off; (void)ws_size; (void)in_sizes; (void)n_in; (void)out_size;

    f32_to_bf16_kernel<<<4096, 256, 0, stream>>>(wl, wl_bf16, (long)K_DIM * N_DIM);

    conv_features_kernel<<<M_DIM, 256, 0, stream>>>(x, w1, b1, w2, b2, act0);

    dim3 grid(N_DIM / 96, M_DIM / 128);   // 67 x 16
    gemm_relu_bf16_kernel<<<grid, 256, 0, stream>>>(act0, wl_bf16, bl, act1);
    gemm_relu_bf16_kernel<<<grid, 256, 0, stream>>>(act1, wl_bf16, bl, act0);
    gemm_relu_bf16_kernel<<<grid, 256, 0, stream>>>(act0, wl_bf16, bl, act1);
    gemm_relu_bf16_kernel<<<grid, 256, 0, stream>>>(act1, wl_bf16, bl, act0);

    head_kernel<<<M_DIM, 256, 0, stream>>>(act0, wo, bo, (float*)d_out);
}
